// GCN_82712480186991
// MI455X (gfx1250) — compile-verified, matrix operand order checked by reading, not run on hardware
//
#include <hip/hip_runtime.h>
#include <hip/hip_bf16.h>

// ---------------------------------------------------------------------------
// GCN (3x GCNConv + PReLU) for gfx1250 / MI455X.
// GEMM: v_wmma_f32_16x16x32_bf16, double-buffered LDS staging via
//       global_load_async_to_lds_b128 (ASYNCcnt-tracked, no VGPR bounce).
// Scatter: hardware global_atomic_add_f32.
// ---------------------------------------------------------------------------

typedef __attribute__((ext_vector_type(16))) __bf16 v16bf;
typedef __attribute__((ext_vector_type(8)))  float  v8f;
typedef int v4i __attribute__((vector_size(16)));           // matches builtin V4i
typedef __attribute__((address_space(1))) v4i* gptr_b128;   // global int4*
typedef __attribute__((address_space(3))) v4i* lptr_b128;   // LDS int4*

#define N_NODES 10000
#define N_EDGES 160000
#define HIDDIM  512
#define M_PAD   10112      // 79 tiles of 128 rows
#define BM      128
#define BN      64
#define BK      32
#define NKSTEP  (HIDDIM / BK)   // 16
#define LSTR    40         // LDS row stride in bf16 (32 data + 8 pad, 16B aligned)

#if defined(__has_builtin)
#  if __has_builtin(__builtin_amdgcn_global_load_async_to_lds_b128)
#    define HAVE_ASYNC_LDS 1
#  endif
#endif
#ifndef HAVE_ASYNC_LDS
#  define HAVE_ASYNC_LDS 0
#endif

// 16-byte global->LDS copy; async (ASYNCcnt-tracked) if available
__device__ __forceinline__ void cp16(const __bf16* g, __bf16* l) {
#if HAVE_ASYNC_LDS
    __builtin_amdgcn_global_load_async_to_lds_b128(
        (gptr_b128)(__bf16*)g,
        (lptr_b128)l,
        0, 0);
#else
    *reinterpret_cast<uint4*>(l) = *reinterpret_cast<const uint4*>(g);
#endif
}

template <int N>
__device__ __forceinline__ void wait_async() {
#if HAVE_ASYNC_LDS
#  if __has_builtin(__builtin_amdgcn_s_wait_asynccnt)
    __builtin_amdgcn_s_wait_asynccnt(N);
#  else
    asm volatile("s_wait_asynccnt %0" :: "i"(N) : "memory");
#  endif
#endif
}

// ---------------- degree / norm ----------------
__global__ void k_deg_init(float* __restrict__ deg) {
    int i = blockIdx.x * 256 + threadIdx.x;
    if (i < N_NODES) deg[i] = 1.0f;            // self loop
}

__global__ void k_deg_edge(float* __restrict__ deg, const int* __restrict__ ei) {
    int e = blockIdx.x * 256 + threadIdx.x;    // grid sized exactly E
    int dst = ei[N_EDGES + e];
    if ((unsigned)dst < (unsigned)N_NODES)
        unsafeAtomicAdd(&deg[dst], 1.0f);
}

__global__ void k_rsqrt(float* __restrict__ deg) {
    int i = blockIdx.x * 256 + threadIdx.x;
    if (i < N_NODES) deg[i] = __frsqrt_rn(deg[i]);   // deg >= 1 always
}

// ---------------- conversions ----------------
// activations f32 [N_NODES,512] -> bf16 [M_PAD,512], tail rows zeroed
__global__ void k_x2bf(const float* __restrict__ cur, __bf16* __restrict__ xb) {
    int i = blockIdx.x * 256 + threadIdx.x;    // grid covers M_PAD*512 exactly
    int row = i >> 9;
    float v = (row < N_NODES) ? cur[i] : 0.0f;
    xb[i] = (__bf16)v;
}

// W f32 [k][n] -> bf16 transposed [n][k]
__global__ void k_w2bf_t(const float* __restrict__ W, __bf16* __restrict__ Wt) {
    int i = blockIdx.x * 256 + threadIdx.x;    // 512*512 exactly
    int n = i & 511, k = i >> 9;
    Wt[(size_t)n * HIDDIM + k] = (__bf16)W[(size_t)k * HIDDIM + n];
}

// ---------------- WMMA GEMM: C[M_PAD,512] = A[M_PAD,512] * B[512,512] ------
// A: bf16 row-major.  Bt: bf16 [n][k] (i.e. W transposed).  C: f32 row-major.
// 256 threads = 8 wave32; each wave owns a 16x64 strip (4x 16x16 f32 accum).
__global__ void __launch_bounds__(256, 2)
k_gemm_bf16(const __bf16* __restrict__ A, const __bf16* __restrict__ Bt,
            float* __restrict__ C) {
    __shared__ __align__(16) __bf16 As[2][BM * LSTR];
    __shared__ __align__(16) __bf16 Bs[2][BN * LSTR];

    const int m0   = blockIdx.x * BM;
    const int n0   = blockIdx.y * BN;
    const int t    = threadIdx.x;
    const int lane = t & 31;
    const int wave = t >> 5;        // 0..7 -> 16-row slice of the 128-row tile
    const int half = lane >> 4;     // 0/1  -> K-interleave half (ISA 7.12.2)
    const int lr   = lane & 15;

    // staging chunk map: thread t covers A chunks (row, row+64) and B chunk (col=row)
    const int rowA = t >> 2;        // 0..63
    const int sub  = t & 3;         // 16B sub-chunk within a 32-element K slice
    const __bf16* gA0 = &A [(size_t)(m0 + rowA)      * HIDDIM + sub * 8];
    const __bf16* gA1 = &A [(size_t)(m0 + rowA + 64) * HIDDIM + sub * 8];
    const __bf16* gB  = &Bt[(size_t)(n0 + rowA)      * HIDDIM + sub * 8];
    const int lA0 = rowA * LSTR + sub * 8;
    const int lA1 = (rowA + 64) * LSTR + sub * 8;

    auto issue = [&](int buf, int k0) {
        cp16(gA0 + k0, &As[buf][lA0]);
        cp16(gA1 + k0, &As[buf][lA1]);
        cp16(gB  + k0, &Bs[buf][lA0]);
    };

    v8f acc0 = {}, acc1 = {}, acc2 = {}, acc3 = {};

    // fragment offsets (ISA 7.12.2 16-bit 16x32 layout):
    // lane<16: K{0..7} + K{16..23}; lane>=16: K{8..15} + K{24..31}
    const int aoff = (wave * 16 + lr) * LSTR + half * 8;
    const int boff = lr * LSTR + half * 8;

    union FragU { uint4 u[2]; v16bf v; };

    issue(0, 0);   // prologue: tile 0 in flight

    for (int i = 0; i < NKSTEP; ++i) {
        const int cur = i & 1;
        if (i + 1 < NKSTEP) {
            issue(cur ^ 1, (i + 1) * BK);   // next tile into other buffer
            wait_async<3>();                // tile i landed (in-order), tile i+1 in flight
        } else {
            wait_async<0>();
        }
        __syncthreads();                    // tile i visible from all waves

        FragU fa;
        fa.u[0] = *reinterpret_cast<const uint4*>(&As[cur][aoff]);
        fa.u[1] = *reinterpret_cast<const uint4*>(&As[cur][aoff + 16]);

        FragU fb0, fb1, fb2, fb3;
        fb0.u[0] = *reinterpret_cast<const uint4*>(&Bs[cur][boff + 0 * 16 * LSTR]);
        fb0.u[1] = *reinterpret_cast<const uint4*>(&Bs[cur][boff + 0 * 16 * LSTR + 16]);
        fb1.u[0] = *reinterpret_cast<const uint4*>(&Bs[cur][boff + 1 * 16 * LSTR]);
        fb1.u[1] = *reinterpret_cast<const uint4*>(&Bs[cur][boff + 1 * 16 * LSTR + 16]);
        fb2.u[0] = *reinterpret_cast<const uint4*>(&Bs[cur][boff + 2 * 16 * LSTR]);
        fb2.u[1] = *reinterpret_cast<const uint4*>(&Bs[cur][boff + 2 * 16 * LSTR + 16]);
        fb3.u[0] = *reinterpret_cast<const uint4*>(&Bs[cur][boff + 3 * 16 * LSTR]);
        fb3.u[1] = *reinterpret_cast<const uint4*>(&Bs[cur][boff + 3 * 16 * LSTR + 16]);

        acc0 = __builtin_amdgcn_wmma_f32_16x16x32_bf16(false, fa.v, false, fb0.v, (short)0, acc0, false, false);
        acc1 = __builtin_amdgcn_wmma_f32_16x16x32_bf16(false, fa.v, false, fb1.v, (short)0, acc1, false, false);
        acc2 = __builtin_amdgcn_wmma_f32_16x16x32_bf16(false, fa.v, false, fb2.v, (short)0, acc2, false, false);
        acc3 = __builtin_amdgcn_wmma_f32_16x16x32_bf16(false, fa.v, false, fb3.v, (short)0, acc3, false, false);

        __syncthreads();                    // reads of buf[cur] done -> safe to refill
    }

    // store: C/D layout = VGPR i holds row (half*8+i), col = lr
    #pragma unroll
    for (int i = 0; i < 8; ++i) {
        size_t r = (size_t)(m0 + wave * 16 + half * 8 + i) * HIDDIM + n0;
        C[r +  0 + lr] = acc0[i];
        C[r + 16 + lr] = acc1[i];
        C[r + 32 + lr] = acc2[i];
        C[r + 48 + lr] = acc3[i];
    }
}

// ---------------- aggregation ----------------
// agg[i,f] = h[i,f]*dinv[i]^2 + b[f]   (self-loop term + bias, also inits agg)
__global__ void k_agg_init(const float* __restrict__ h, const float* __restrict__ dinv,
                           const float* __restrict__ bias, float* __restrict__ agg) {
    int tid = blockIdx.x * 256 + threadIdx.x;   // N_NODES*128 exactly
    int i = tid >> 7;
    int f = (tid & 127) << 2;
    float s = dinv[i]; s = s * s;
    float4 v = *reinterpret_cast<const float4*>(&h[(size_t)i * HIDDIM + f]);
    float4 bb = *reinterpret_cast<const float4*>(&bias[f]);
    float4 o = { v.x * s + bb.x, v.y * s + bb.y, v.z * s + bb.z, v.w * s + bb.w };
    *reinterpret_cast<float4*>(&agg[(size_t)i * HIDDIM + f]) = o;
}

// per-edge gather + scatter-add: agg[dst] += h[src] * dinv[src]*dinv[dst]
__global__ void __launch_bounds__(256) k_scatter(const float* __restrict__ h,
                                                 const float* __restrict__ dinv,
                                                 const int* __restrict__ ei,
                                                 float* __restrict__ agg) {
    int tid = blockIdx.x * 256 + threadIdx.x;   // N_EDGES*128 exactly
    int e = tid >> 7;
    int f = (tid & 127) << 2;
    int src = ei[e];
    int dst = ei[N_EDGES + e];
    if ((unsigned)src >= (unsigned)N_NODES || (unsigned)dst >= (unsigned)N_NODES) return;
    float nrm = dinv[src] * dinv[dst];
    float4 v = *reinterpret_cast<const float4*>(&h[(size_t)src * HIDDIM + f]);
    float* p = &agg[(size_t)dst * HIDDIM + f];
    unsafeAtomicAdd(p + 0, v.x * nrm);
    unsafeAtomicAdd(p + 1, v.y * nrm);
    unsafeAtomicAdd(p + 2, v.z * nrm);
    unsafeAtomicAdd(p + 3, v.w * nrm);
}

__global__ void k_prelu(const float* __restrict__ agg, const float* __restrict__ a,
                        float* __restrict__ out) {
    int i = blockIdx.x * 256 + threadIdx.x;     // N_NODES*512 exactly
    int c = i & 511;
    float v = agg[i];
    out[i] = (v >= 0.0f) ? v : a[c] * v;
}

// ---------------------------------------------------------------------------
static inline size_t align256(size_t x) { return (x + 255) & ~(size_t)255; }

extern "C" void kernel_launch(void* const* d_in, const int* in_sizes, int n_in,
                              void* d_out, int out_size, void* d_ws, size_t ws_size,
                              hipStream_t stream) {
    const float* x  = (const float*)d_in[0];
    const int*   ei = (const int*)d_in[1];
    const float* W[3] = {(const float*)d_in[2], (const float*)d_in[5], (const float*)d_in[8]};
    const float* b[3] = {(const float*)d_in[3], (const float*)d_in[6], (const float*)d_in[9]};
    const float* a[3] = {(const float*)d_in[4], (const float*)d_in[7], (const float*)d_in[10]};

    char* ws = (char*)d_ws;
    float*  dinv = (float*)ws;  ws += align256((size_t)N_NODES * 4);
    __bf16* xb   = (__bf16*)ws; ws += align256((size_t)M_PAD * HIDDIM * 2);
    __bf16* Wt   = (__bf16*)ws; ws += align256((size_t)HIDDIM * HIDDIM * 2);
    float*  h    = (float*)ws;  ws += align256((size_t)M_PAD * HIDDIM * 4);
    float*  agg  = (float*)ws;  ws += align256((size_t)N_NODES * HIDDIM * 4);

    // normalization (edge-structure dependent only, once per call)
    k_deg_init<<<(N_NODES + 255) / 256, 256, 0, stream>>>(dinv);
    k_deg_edge<<<N_EDGES / 256, 256, 0, stream>>>(dinv, ei);
    k_rsqrt<<<(N_NODES + 255) / 256, 256, 0, stream>>>(dinv);

    for (int l = 0; l < 3; ++l) {
        const float* cur = (l == 0) ? x : agg;
        k_x2bf<<<(M_PAD * HIDDIM) / 256, 256, 0, stream>>>(cur, xb);
        k_w2bf_t<<<(HIDDIM * HIDDIM) / 256, 256, 0, stream>>>(W[l], Wt);

        dim3 g(M_PAD / BM, HIDDIM / BN);   // 79 x 8
        k_gemm_bf16<<<g, 256, 0, stream>>>(xb, Wt, h);

        k_agg_init<<<(N_NODES * 128) / 256, 256, 0, stream>>>(h, dinv, b[l], agg);
        k_scatter<<<(N_EDGES * 128) / 256, 256, 0, stream>>>(h, dinv, ei, agg);

        float* out = (l == 2) ? (float*)d_out : agg;
        k_prelu<<<(N_NODES * HIDDIM) / 256, 256, 0, stream>>>(agg, a[l], out);
    }
}
